// QNet_77635828843174
// MI455X (gfx1250) — matrix operand order repaired and checked by reading, not compiled, not run on hardware
//
#include <hip/hip_runtime.h>
#include <hip/hip_bf16.h>

typedef __attribute__((ext_vector_type(16))) _Float16 v16h;
typedef __attribute__((ext_vector_type(8)))  float    v8f;

#define B_   8
#define L_   4096
#define DIN_ 1024
#define H_   1024
#define LAT_ 256
#define SEG_ 128
#define K_   32

// ---------------------------------------------------------------------------
// Stage 1: fused projection + decay-weighted per-segment reduction.
// Block = 512 threads (16 wave32). Grid = (H/128, K, B).
// - Reads raw fp32 inputs once (no separate conversion pass): f32 -> f16
//   conversion happens in registers during LDS staging.
// - Double-buffered LDS + register prefetch of the next K-chunk so global
//   latency overlaps the WMMA chain.
// - Wave w owns stream (w>>3) and M-strip (w&7): 8 v8f accumulators per wave.
//   Ccorr[h] = sum_j a^{127-j} * (targets[s+j,:] @ Wp[:,h])
//   D[h]     = sum_j a^{127-j} * (decoder[s+j,:] @ Wp[:,h])
// ---------------------------------------------------------------------------
__global__ void __launch_bounds__(512, 1)
seg_proj_kernel(const float* __restrict__ Xc,   // targets f32 (B*L, DIN)
                const float* __restrict__ Xd,   // decoder f32 (B*L, DIN)
                const float* __restrict__ Wp,   // W_proj f32 (DIN, H)
                const float* __restrict__ a_raw,
                float* __restrict__ Cg,         // (B,K,H)
                float* __restrict__ Dg) {       // (B,K,H)
  __shared__ alignas(16) _Float16 Ac[2][128][40];  // targets chunk (80B row stride)
  __shared__ alignas(16) _Float16 Ad[2][128][40];  // decoder chunk
  __shared__ alignas(16) _Float16 Bt[2][128][40];  // W tile transposed: Bt[n][kk]
  __shared__ float redC[128];
  __shared__ float redD[128];

  const int tid   = threadIdx.x;
  const int hbase = blockIdx.x * 128;
  const int k     = blockIdx.y;
  const int b     = blockIdx.z;
  const int lane  = tid & 31;
  const int wv    = tid >> 5;        // 0..15
  const int strip = wv & 7;          // M strip [strip*16, strip*16+16)
  const int strm  = wv >> 3;         // 0 = corr (targets), 1 = dec
  const int halfg = lane >> 4;
  const int mn    = lane & 15;

  if (tid < 128) redC[tid] = 0.0f;
  else if (tid < 256) redD[tid - 128] = 0.0f;

  v8f acc[8];
  const v8f vzero = {0.f,0.f,0.f,0.f,0.f,0.f,0.f,0.f};
#pragma unroll
  for (int i = 0; i < 8; i++) acc[i] = vzero;

  const long rowbase = (long)b * L_ + (long)k * SEG_;

  // Per-thread staging assignments (fixed across chunks):
  const int xrow = tid >> 2, xq = tid & 3;        // X: 8 floats per stream
  const int wkk  = tid >> 4, wg = tid & 15;       // W: 8 floats
  const long xoff = (rowbase + xrow) * (long)DIN_ + xq * 8;
  const long woff = (long)wkk * H_ + hbase + wg * 8;

  const float4 f4z = {0.f, 0.f, 0.f, 0.f};
  float4 cxc0, cxc1, cxd0, cxd1, cw0, cw1;        // current chunk (f32)
  float4 nxc0 = f4z, nxc1 = f4z, nxd0 = f4z, nxd1 = f4z, nw0 = f4z, nw1 = f4z;

  // Prologue: load chunk 0.
  {
    const float4* pc = reinterpret_cast<const float4*>(Xc + xoff);
    cxc0 = pc[0]; cxc1 = pc[1];
    const float4* pd = reinterpret_cast<const float4*>(Xd + xoff);
    cxd0 = pd[0]; cxd1 = pd[1];
    const float4* pw = reinterpret_cast<const float4*>(Wp + woff);
    cw0 = pw[0]; cw1 = pw[1];
  }

  int p = 0;
  for (int kb = 0; kb < DIN_; kb += 32, p ^= 1) {
    // Prefetch next chunk into registers (overlaps with staging + compute).
    if (kb + 32 < DIN_) {
      const float4* pc = reinterpret_cast<const float4*>(Xc + xoff + kb + 32);
      nxc0 = pc[0]; nxc1 = pc[1];
      const float4* pd = reinterpret_cast<const float4*>(Xd + xoff + kb + 32);
      nxd0 = pd[0]; nxd1 = pd[1];
      const float4* pw = reinterpret_cast<const float4*>(Wp + woff + (long)(kb + 32) * H_);
      nw0 = pw[0]; nw1 = pw[1];
    }

    // Stage current chunk into LDS buffer p, converting f32 -> f16 in regs.
    {
      union { uint4 q4; _Float16 h[8]; } hc, hd;
      hc.h[0] = (_Float16)cxc0.x; hc.h[1] = (_Float16)cxc0.y;
      hc.h[2] = (_Float16)cxc0.z; hc.h[3] = (_Float16)cxc0.w;
      hc.h[4] = (_Float16)cxc1.x; hc.h[5] = (_Float16)cxc1.y;
      hc.h[6] = (_Float16)cxc1.z; hc.h[7] = (_Float16)cxc1.w;
      hd.h[0] = (_Float16)cxd0.x; hd.h[1] = (_Float16)cxd0.y;
      hd.h[2] = (_Float16)cxd0.z; hd.h[3] = (_Float16)cxd0.w;
      hd.h[4] = (_Float16)cxd1.x; hd.h[5] = (_Float16)cxd1.y;
      hd.h[6] = (_Float16)cxd1.z; hd.h[7] = (_Float16)cxd1.w;
      *reinterpret_cast<uint4*>(&Ac[p][xrow][xq * 8]) = hc.q4;
      *reinterpret_cast<uint4*>(&Ad[p][xrow][xq * 8]) = hd.q4;
      _Float16 wh[8];
      wh[0] = (_Float16)cw0.x; wh[1] = (_Float16)cw0.y;
      wh[2] = (_Float16)cw0.z; wh[3] = (_Float16)cw0.w;
      wh[4] = (_Float16)cw1.x; wh[5] = (_Float16)cw1.y;
      wh[6] = (_Float16)cw1.z; wh[7] = (_Float16)cw1.w;
#pragma unroll
      for (int i = 0; i < 8; i++) Bt[p][wg * 8 + i][wkk] = wh[i];
    }
    __syncthreads();

    // Compute on buffer p.
    const _Float16 (*At)[40] = (strm == 0) ? Ac[p] : Ad[p];   // wave-uniform
    // A fragment: half-group 0 holds K {0..15},
    // half-group 1 holds K {16..31}; contiguous, matching the B fragment.
    union FA { v16h v; unsigned u[8]; } fa;
#pragma unroll
    for (int q = 0; q < 8; q++) {
      int kk = halfg * 16 + (q << 1);
      fa.u[q] = *reinterpret_cast<const unsigned*>(&At[strip * 16 + mn][kk]);
    }
#pragma unroll
    for (int nt = 0; nt < 8; nt++) {
      // B fragment: lane column = mn, K = halfg*16 + 0..15 contiguous.
      union FB { v16h v; uint4 q[2]; } fb;
      const _Float16* bp = &Bt[p][nt * 16 + mn][halfg * 16];
      fb.q[0] = *reinterpret_cast<const uint4*>(bp);
      fb.q[1] = *reinterpret_cast<const uint4*>(bp + 8);
      acc[nt] = __builtin_amdgcn_wmma_f32_16x16x32_f16(false, fa.v, false, fb.v,
                                                       (short)0, acc[nt], false, false);
    }
    __syncthreads();

    // Rotate prefetched registers into "current".
    cxc0 = nxc0; cxc1 = nxc1; cxd0 = nxd0; cxd1 = nxd1; cw0 = nw0; cw1 = nw1;
  }

  // Epilogue: weight row j by a^{127-j}, reduce 128 rows into redC/redD.
  // C layout: vgpr r, lane -> (M = r + 8*halfg, N = mn).
  float* red = (strm == 0) ? redC : redD;   // wave-uniform
#pragma unroll
  for (int nt = 0; nt < 8; nt++) {
    int hl = nt * 16 + mn;
    float ar = a_raw[hbase + hl];
    float a  = 1.0f / (1.0f + expf(-ar));      // sigmoid
    float la = log2f(a);
    float pr = 0.f;
#pragma unroll
    for (int r = 0; r < 8; r++) {
      int j = strip * 16 + halfg * 8 + r;      // segment position
      float w = exp2f(la * (float)(127 - j));  // a^{127-j}
      pr += w * acc[nt][r];
    }
    atomicAdd(&red[hl], pr);                   // ds_add_f32
  }
  __syncthreads();

  long outb = ((long)b * K_ + k) * (long)H_ + hbase;
  if (tid < 128) Cg[outb + tid] = redC[tid];
  else if (tid < 256) Dg[outb + (tid - 128)] = redD[tid - 128];
}

// ---------------------------------------------------------------------------
// Stage 2: per-(b,h) scan over K=32 segments.
// hT_k = a^128 * carry + D_k ; carry <- a^128 * carry + Ccorr_k
// b_proj folded via geometric sum (1-a^128)/(1-a).
// ---------------------------------------------------------------------------
__global__ void scan_kernel(const float* __restrict__ Cg, const float* __restrict__ Dg,
                            const float* __restrict__ a_raw, const float* __restrict__ b_proj,
                            _Float16* __restrict__ hT16) {
  int idx = blockIdx.x * blockDim.x + threadIdx.x;  // 0 .. B*H-1
  int b = idx >> 10;
  int h = idx & (H_ - 1);
  float a  = 1.0f / (1.0f + expf(-a_raw[h]));
  float la = log2f(a);
  float aP = exp2f(la * 128.0f);                    // a^SEG
  float geo = (fabsf(1.0f - a) < 1e-6f) ? 128.0f : (1.0f - aP) / (1.0f - a);
  float bp = b_proj[h] * geo;
  float carry = 0.0f;
  for (int k = 0; k < K_; k++) {
    long o = ((long)b * K_ + k) * (long)H_ + h;
    float hv = aP * carry + (Dg[o] + bp);
    carry    = aP * carry + (Cg[o] + bp);
    hT16[o]  = (_Float16)hv;
  }
}

// ---------------------------------------------------------------------------
// Stages 3/4: generic 128x128-tile WMMA GEMM. A is f16 (produced internally),
// B is the ORIGINAL fp32 weight matrix (converted during LDS staging), + bias,
// optional SiLU; output to f16 (intermediate y) or f32 (heads into d_out).
// Grid = (N/128, ceil(M/128)), block = 256 (8 wave32, 8 accs/wave).
// ---------------------------------------------------------------------------
__global__ void __launch_bounds__(256, 1)
gemm_kernel(const _Float16* __restrict__ A,   // (M, Kd) row-major f16
            const float* __restrict__ Bm,     // (Kd, N) row-major f32
            const float* __restrict__ bias,   // (N,)
            float* __restrict__ outF32,
            _Float16* __restrict__ outF16,
            int M, int N, int Kd, int act) {
  __shared__ alignas(16) _Float16 As[128][40];
  __shared__ alignas(16) _Float16 Bt[128][40];
  const int tid   = threadIdx.x;
  const int nbase = blockIdx.x * 128;
  const int mbase = blockIdx.y * 128;
  const int lane  = tid & 31;
  const int wv    = tid >> 5;
  const int halfg = lane >> 4;
  const int mn    = lane & 15;

  v8f acc[8];
  const v8f vzero = {0.f,0.f,0.f,0.f,0.f,0.f,0.f,0.f};
#pragma unroll
  for (int i = 0; i < 8; i++) acc[i] = vzero;

  for (int kb = 0; kb < Kd; kb += 32) {
    __syncthreads();
    for (int c = tid; c < 512; c += 256) {
      int row = c >> 2, q = c & 3;
      *reinterpret_cast<uint4*>(&As[row][q * 8]) =
          *reinterpret_cast<const uint4*>(A + (long)(mbase + row) * Kd + kb + q * 8);
    }
    {
      int kk = tid >> 3, g = tid & 7;
      const float4* wsrc = reinterpret_cast<const float4*>(Bm + (long)(kb + kk) * N + nbase + g * 16);
      float4 f0 = wsrc[0], f1 = wsrc[1], f2 = wsrc[2], f3 = wsrc[3];
      _Float16 h[16];
      h[0]=(_Float16)f0.x;  h[1]=(_Float16)f0.y;  h[2]=(_Float16)f0.z;  h[3]=(_Float16)f0.w;
      h[4]=(_Float16)f1.x;  h[5]=(_Float16)f1.y;  h[6]=(_Float16)f1.z;  h[7]=(_Float16)f1.w;
      h[8]=(_Float16)f2.x;  h[9]=(_Float16)f2.y;  h[10]=(_Float16)f2.z; h[11]=(_Float16)f2.w;
      h[12]=(_Float16)f3.x; h[13]=(_Float16)f3.y; h[14]=(_Float16)f3.z; h[15]=(_Float16)f3.w;
#pragma unroll
      for (int i = 0; i < 16; i++) Bt[g * 16 + i][kk] = h[i];
    }
    __syncthreads();

    union FA { v16h v; unsigned u[8]; } fa;
#pragma unroll
    for (int q = 0; q < 8; q++) {
      int kk = halfg * 16 + (q << 1);
      fa.u[q] = *reinterpret_cast<const unsigned*>(&As[wv * 16 + mn][kk]);
    }
#pragma unroll
    for (int nt = 0; nt < 8; nt++) {
      union FB { v16h v; uint4 q[2]; } fb;
      const _Float16* bp = &Bt[nt * 16 + mn][halfg * 16];
      fb.q[0] = *reinterpret_cast<const uint4*>(bp);
      fb.q[1] = *reinterpret_cast<const uint4*>(bp + 8);
      acc[nt] = __builtin_amdgcn_wmma_f32_16x16x32_f16(false, fa.v, false, fb.v,
                                                       (short)0, acc[nt], false, false);
    }
  }

#pragma unroll
  for (int nt = 0; nt < 8; nt++) {
    int col = nbase + nt * 16 + mn;
    float bv = bias[col];
#pragma unroll
    for (int r = 0; r < 8; r++) {
      int row = mbase + wv * 16 + halfg * 8 + r;
      if (row >= M) continue;
      float v = acc[nt][r] + bv;
      if (act) v = v / (1.0f + expf(-v));   // SiLU
      if (outF16) outF16[(long)row * N + col] = (_Float16)v;
      else        outF32[(long)row * N + col] = v;
    }
  }
}

// ---------------------------------------------------------------------------
// Host launcher
// ---------------------------------------------------------------------------
extern "C" void kernel_launch(void* const* d_in, const int* in_sizes, int n_in,
                              void* d_out, int out_size, void* d_ws, size_t ws_size,
                              hipStream_t stream) {
  const float* dec    = (const float*)d_in[0];   // decoder_output (B,L,DIN)
  const float* tgt    = (const float*)d_in[1];   // targets        (B,L,DIN)
  const float* W_proj = (const float*)d_in[2];   // (DIN,H)
  const float* b_proj = (const float*)d_in[3];   // (H,)
  const float* a_raw  = (const float*)d_in[4];   // (H,)
  const float* W_out  = (const float*)d_in[5];   // (H,H)
  const float* b_out  = (const float*)d_in[6];   // (H,)
  const float* W_mu   = (const float*)d_in[7];   // (H,LAT)
  const float* b_mu   = (const float*)d_in[8];   // (LAT,)
  const float* W_lv   = (const float*)d_in[9];   // (H,LAT)
  const float* b_lv   = (const float*)d_in[10];  // (LAT,)
  float* out = (float*)d_out;                    // mu (B,K,LAT) then logvar (B,K,LAT)

  char* ws = (char*)d_ws;
  size_t off = 0;
  auto alloc = [&](size_t bytes) {
    char* p = ws + off;
    off += (bytes + 255) & ~(size_t)255;
    return p;
  };
  float*    Cg   = (float*)   alloc((size_t)B_ * K_ * H_ * 4);
  float*    Dg   = (float*)   alloc((size_t)B_ * K_ * H_ * 4);
  _Float16* hT16 = (_Float16*)alloc((size_t)B_ * K_ * H_ * 2);
  _Float16* y16  = (_Float16*)alloc((size_t)B_ * K_ * H_ * 2);

  seg_proj_kernel<<<dim3(H_ / 128, K_, B_), 512, 0, stream>>>(tgt, dec, W_proj, a_raw, Cg, Dg);

  scan_kernel<<<dim3((B_ * H_) / 256), 256, 0, stream>>>(Cg, Dg, a_raw, b_proj, hT16);

  // y = silu(hT @ W_out + b_out)  : (256,1024) x (1024,1024)
  gemm_kernel<<<dim3(H_ / 128, (B_ * K_) / 128), 256, 0, stream>>>(
      hT16, W_out, b_out, nullptr, y16, B_ * K_, H_, H_, 1);
  // mu = y @ W_mu + b_mu ; logvar = y @ W_lv + b_lv  : (256,1024) x (1024,256)
  gemm_kernel<<<dim3(LAT_ / 128, (B_ * K_) / 128), 256, 0, stream>>>(
      y16, W_mu, b_mu, out, nullptr, B_ * K_, LAT_, H_, 0);
  gemm_kernel<<<dim3(LAT_ / 128, (B_ * K_) / 128), 256, 0, stream>>>(
      y16, W_lv, b_lv, out + (long)B_ * K_ * LAT_, nullptr, B_ * K_, LAT_, H_, 0);
}
